// SelfAttention_17102559773267
// MI455X (gfx1250) — matrix operand order repaired and checked by reading, not compiled, not run on hardware
//
#include <hip/hip_runtime.h>

// ---------------------------------------------------------------------------
// CDNA5 (gfx1250) bf16 WMMA self-attention pipeline
//   shapes: B=4, S=2048, D=1024, H=16, hd=64  -> M = B*S = 8192
// ---------------------------------------------------------------------------

typedef __attribute__((ext_vector_type(16))) __bf16 v16bf;
typedef __attribute__((ext_vector_type(8)))  float  v8f;

union Frag {
  uint4 u[2];
  v16bf v;
};

__device__ __forceinline__ unsigned short f32_to_bf16(float f) {
  unsigned int u = __float_as_uint(f);
  u += 0x7FFFu + ((u >> 16) & 1u);   // round-to-nearest-even
  return (unsigned short)(u >> 16);
}

#define WMMA_BF16(a, b, c) \
  __builtin_amdgcn_wmma_f32_16x16x32_bf16(false, (a), false, (b), (short)0, (c), false, false)

// ---------------------------------------------------------------------------
// Kernel 1: f32 -> bf16 elementwise convert (x)
// ---------------------------------------------------------------------------
__global__ __launch_bounds__(256) void cvt_f32_bf16_kernel(
    const float* __restrict__ in, unsigned short* __restrict__ out, int n) {
  int i = (blockIdx.x * 256 + threadIdx.x) * 4;
  if (i + 3 < n) {
    float4 f = *(const float4*)(in + i);
    ushort4 o;
    o.x = f32_to_bf16(f.x);
    o.y = f32_to_bf16(f.y);
    o.z = f32_to_bf16(f.z);
    o.w = f32_to_bf16(f.w);
    *(ushort4*)(out + i) = o;
  }
}

// ---------------------------------------------------------------------------
// Kernel 2: transpose-convert weights: in [K][N] f32 -> out [N][K] bf16
// ---------------------------------------------------------------------------
__global__ __launch_bounds__(256) void transpose_cvt_kernel(
    const float* __restrict__ in, unsigned short* __restrict__ out, int K, int N) {
  __shared__ float tile[32][33];
  const int n0 = blockIdx.x * 32, k0 = blockIdx.y * 32;
  const int tx = threadIdx.x, ty = threadIdx.y;  // block (32, 8)
#pragma unroll
  for (int r = 0; r < 32; r += 8)
    tile[ty + r][tx] = in[(size_t)(k0 + ty + r) * N + n0 + tx];
  __syncthreads();
#pragma unroll
  for (int r = 0; r < 32; r += 8)
    out[(size_t)(n0 + ty + r) * K + k0 + tx] = f32_to_bf16(tile[tx][ty + r]);
}

// ---------------------------------------------------------------------------
// Kernel 3/5: bf16 GEMM  C = A[M][K] * Bt[N][K]^T + bias
//   block tile 128x128, 256 threads = 8 waves (4 m-strips x 2 n-strips),
//   each wave 32(M) x 64(N) = 8 WMMAs/k-step from 2 A-frags + 4 B-frags.
//   LDS double-buffered: next k-tile's global loads overlap the WMMA chain.
//   mode 0: scatter into Q (scaled 1/8), K, Vt (bf16, head-split)
//   mode 1: f32 output [M][N]
// ---------------------------------------------------------------------------
__global__ __launch_bounds__(256) void gemm_bf16_kernel(
    const unsigned short* __restrict__ A,
    const unsigned short* __restrict__ Bt,
    const float* __restrict__ bias,
    int K, int mode,
    unsigned short* __restrict__ Qb,
    unsigned short* __restrict__ Kb,
    unsigned short* __restrict__ Vtb,
    float* __restrict__ Cout, int N) {
  __shared__ unsigned short sA[2][128 * 32];
  __shared__ unsigned short sB[2][128 * 32];
  const int tid  = threadIdx.x;
  const int wave = tid >> 5, lane = tid & 31;
  const int half = lane >> 4, l16 = lane & 15;
  const int wm = wave & 3;   // 4 m-strips of 32 rows
  const int wn = wave >> 2;  // 2 n-strips of 64 cols
  const int m0 = blockIdx.x * 128, n0 = blockIdx.y * 128;

  const int r = tid >> 1;          // 0..127 (tile row)
  const int h = (tid & 1) * 16;    // 16-element half of the 32-wide k-slice
  const unsigned short* gA = A + (size_t)(m0 + r) * K + h;
  const unsigned short* gB = Bt + (size_t)(n0 + r) * K + h;

  v8f acc[2][4];
#pragma unroll
  for (int tm = 0; tm < 2; ++tm)
#pragma unroll
    for (int t = 0; t < 4; ++t)
#pragma unroll
      for (int i = 0; i < 8; ++i) acc[tm][t][i] = 0.f;

  // prologue: tile 0 -> buffer 0
  {
    uint4 a0 = ((const uint4*)gA)[0], a1 = ((const uint4*)gA)[1];
    uint4 b0 = ((const uint4*)gB)[0], b1 = ((const uint4*)gB)[1];
    uint4* pa = (uint4*)(sA[0] + r * 32 + h);
    pa[0] = a0; pa[1] = a1;
    uint4* pb = (uint4*)(sB[0] + r * 32 + h);
    pb[0] = b0; pb[1] = b1;
  }

  const int nk = K >> 5;
  int cur = 0;
  for (int it = 0; it < nk; ++it) {
    __syncthreads();
    const bool pf = (it + 1) < nk;
    uint4 a0 = {}, a1 = {}, b0 = {}, b1 = {};
    if (pf) {  // issue next tile's loads; they overlap the WMMA chain below
      const uint4* ga = (const uint4*)(gA + (it + 1) * 32);
      const uint4* gb = (const uint4*)(gB + (it + 1) * 32);
      a0 = ga[0]; a1 = ga[1];
      b0 = gb[0]; b1 = gb[1];
    }

    Frag af[2], bf[4];
#pragma unroll
    for (int tm = 0; tm < 2; ++tm) {
      const unsigned short* ap = sA[cur] + (wm * 32 + tm * 16 + l16) * 32;
      af[tm].u[0] = *(const uint4*)(ap + half * 8);
      af[tm].u[1] = *(const uint4*)(ap + 16 + half * 8);
    }
#pragma unroll
    for (int t = 0; t < 4; ++t) {
      const unsigned short* bp = sB[cur] + (wn * 64 + t * 16 + l16) * 32 + half * 8;
      bf[t].u[0] = ((const uint4*)bp)[0];
      bf[t].u[1] = ((const uint4*)(bp + 16))[0];
    }
#pragma unroll
    for (int tm = 0; tm < 2; ++tm)
#pragma unroll
      for (int t = 0; t < 4; ++t)
        acc[tm][t] = WMMA_BF16(af[tm].v, bf[t].v, acc[tm][t]);

    if (pf) {
      uint4* pa = (uint4*)(sA[cur ^ 1] + r * 32 + h);
      pa[0] = a0; pa[1] = a1;
      uint4* pb = (uint4*)(sB[cur ^ 1] + r * 32 + h);
      pb[0] = b0; pb[1] = b1;
    }
    cur ^= 1;
  }

  if (mode == 0) {
#pragma unroll
    for (int tm = 0; tm < 2; ++tm) {
#pragma unroll
      for (int t = 0; t < 4; ++t) {
#pragma unroll
        for (int i = 0; i < 8; ++i) {
          const int gm = m0 + wm * 32 + tm * 16 + half * 8 + i;
          const int gn = n0 + wn * 64 + t * 16 + l16;
          const float v = acc[tm][t][i] + bias[gn];
          const int bidx = gm >> 11, s = gm & 2047;
          if (gn < 1024) {
            const int hh = gn >> 6, d = gn & 63;
            Qb[((size_t)(bidx * 16 + hh) * 2048 + s) * 64 + d] = f32_to_bf16(v * 0.125f);
          } else if (gn < 2048) {
            const int nn = gn - 1024, hh = nn >> 6, d = nn & 63;
            Kb[((size_t)(bidx * 16 + hh) * 2048 + s) * 64 + d] = f32_to_bf16(v);
          } else {
            const int nn = gn - 2048, hh = nn >> 6, d = nn & 63;
            Vtb[((size_t)(bidx * 16 + hh) * 64 + d) * 2048 + s] = f32_to_bf16(v);
          }
        }
      }
    }
  } else {
#pragma unroll
    for (int tm = 0; tm < 2; ++tm)
#pragma unroll
      for (int t = 0; t < 4; ++t)
#pragma unroll
        for (int i = 0; i < 8; ++i) {
          const int gm = m0 + wm * 32 + tm * 16 + half * 8 + i;
          const int gn = n0 + wn * 64 + t * 16 + l16;
          Cout[(size_t)gm * N + gn] = acc[tm][t][i] + bias[gn];
        }
  }
}

// ---------------------------------------------------------------------------
// Kernel 4: causal flash attention, one (b,h) x 128-query-row block per WG.
//   8 waves * 16 rows; 64-key LDS tiles of K [key][d] and V^T [d][key].
//   One online-softmax statistics update per 64 keys (4 score tiles),
//   halving shuffle/exp cost per key vs a 32-key update.
//   P routed through per-wave LDS (C-layout -> A-layout), 8 PV WMMAs (k=64).
// ---------------------------------------------------------------------------
__global__ __launch_bounds__(256) void flash_attn_kernel(
    const unsigned short* __restrict__ Qb,
    const unsigned short* __restrict__ Kb,
    const unsigned short* __restrict__ Vtb,
    unsigned short* __restrict__ Ob) {
  const int S = 2048;
  const int bh = blockIdx.x;
  const int b  = bh >> 4;
  const int h  = bh & 15;
  const int qblk = blockIdx.y;
  const int tid  = threadIdx.x;
  const int wave = tid >> 5, lane = tid & 31;
  const int half = lane >> 4, l16 = lane & 15;

  __shared__ unsigned short sK[64 * 64];       // [key][d]
  __shared__ unsigned short sV[64 * 64];       // [d][key]
  __shared__ unsigned short sP[8 * 16 * 64];   // per-wave P staging (16x64)

  const size_t head = (size_t)bh * S * 64;
  const int qbase = qblk * 128 + wave * 16;
  const int qtop  = qbase + 15;

  Frag qf0, qf1;  // Q A-fragments, k-chunks [0,32) and [32,64)
  {
    const unsigned short* qp = Qb + head + (size_t)(qbase + l16) * 64;
    qf0.u[0] = *(const uint4*)(qp + half * 8);
    qf0.u[1] = *(const uint4*)(qp + 16 + half * 8);
    qf1.u[0] = *(const uint4*)(qp + 32 + half * 8);
    qf1.u[1] = *(const uint4*)(qp + 48 + half * 8);
  }

  v8f oacc[4];
  float mrow[8], lrow[8];
#pragma unroll
  for (int t = 0; t < 4; ++t)
#pragma unroll
    for (int i = 0; i < 8; ++i) oacc[t][i] = 0.f;
#pragma unroll
  for (int i = 0; i < 8; ++i) { mrow[i] = -1e30f; lrow[i] = 0.f; }

  const int nkb = 2 * qblk + 2;  // 64-key blocks needed for causality
  for (int kb = 0; kb < nkb; ++kb) {
    {
      const int r = tid >> 2;
      const int c = (tid & 3) * 16;
      const uint4* gk = (const uint4*)(Kb + head + (size_t)(kb * 64 + r) * 64 + c);
      *(uint4*)(sK + r * 64 + c)     = gk[0];
      *(uint4*)(sK + r * 64 + c + 8) = gk[1];
      const uint4* gv = (const uint4*)(Vtb + (size_t)bh * 64 * S + (size_t)r * S + kb * 64 + c);
      *(uint4*)(sV + r * 64 + c)     = gv[0];
      *(uint4*)(sV + r * 64 + c + 8) = gv[1];
    }
    __syncthreads();

    if (kb * 64 <= qtop) {  // skip fully-masked 64-key blocks for this wave
      // ---- scores: 4 x (16x16) tiles, contraction over hd = 64 ----
      v8f sacc[4];
#pragma unroll
      for (int c2 = 0; c2 < 4; ++c2) {
#pragma unroll
        for (int i = 0; i < 8; ++i) sacc[c2][i] = 0.f;
        const unsigned short* kp = sK + (c2 * 16 + l16) * 64;
        Frag kf0, kf1;
        kf0.u[0] = *(const uint4*)(kp + half * 8);
        kf0.u[1] = *(const uint4*)(kp + 16 + half * 8);
        kf1.u[0] = *(const uint4*)(kp + 32 + half * 8);
        kf1.u[1] = *(const uint4*)(kp + 48 + half * 8);
        sacc[c2] = WMMA_BF16(qf0.v, kf0.v, sacc[c2]);
        sacc[c2] = WMMA_BF16(qf1.v, kf1.v, sacc[c2]);
      }

      // ---- causal mask: only the diagonal block needs it ----
      if (kb * 64 + 63 > qtop) {
#pragma unroll
        for (int c2 = 0; c2 < 4; ++c2) {
          const int key = kb * 64 + c2 * 16 + l16;
#pragma unroll
          for (int i = 0; i < 8; ++i) {
            if (key > qbase + half * 8 + i) sacc[c2][i] = -10000.0f;
          }
        }
      }

      // ---- online softmax: one stats update per 64 keys ----
#pragma unroll
      for (int i = 0; i < 8; ++i) {
        float t = fmaxf(fmaxf(sacc[0][i], sacc[1][i]), fmaxf(sacc[2][i], sacc[3][i]));
        t = fmaxf(t, __shfl_xor(t, 1, 32));
        t = fmaxf(t, __shfl_xor(t, 2, 32));
        t = fmaxf(t, __shfl_xor(t, 4, 32));
        t = fmaxf(t, __shfl_xor(t, 8, 32));
        const float mnew = fmaxf(mrow[i], t);
        const float fac  = __expf(mrow[i] - mnew);
        mrow[i] = mnew;
        float ps = 0.f;
#pragma unroll
        for (int c2 = 0; c2 < 4; ++c2) {
          const float p = __expf(sacc[c2][i] - mnew);
          sacc[c2][i] = p;
          ps += p;
        }
        ps += __shfl_xor(ps, 1, 32);
        ps += __shfl_xor(ps, 2, 32);
        ps += __shfl_xor(ps, 4, 32);
        ps += __shfl_xor(ps, 8, 32);
        lrow[i] = lrow[i] * fac + ps;
#pragma unroll
        for (int t2 = 0; t2 < 4; ++t2) oacc[t2][i] *= fac;
      }

      // ---- stage P: C-layout -> LDS -> A-layout (intra-wave, DS in-order) ----
      unsigned short* pw = sP + wave * (16 * 64);
#pragma unroll
      for (int c2 = 0; c2 < 4; ++c2)
#pragma unroll
        for (int i = 0; i < 8; ++i)
          pw[(half * 8 + i) * 64 + c2 * 16 + l16] = f32_to_bf16(sacc[c2][i]);
      asm volatile("s_wait_dscnt 0" ::: "memory");
      Frag pf0, pf1;
      {
        const unsigned short* pr = pw + l16 * 64;
        pf0.u[0] = *(const uint4*)(pr + half * 8);
        pf0.u[1] = *(const uint4*)(pr + 16 + half * 8);
        pf1.u[0] = *(const uint4*)(pr + 32 + half * 8);
        pf1.u[1] = *(const uint4*)(pr + 48 + half * 8);
      }

      // ---- O += P @ V : contraction over 64 keys ----
#pragma unroll
      for (int t2 = 0; t2 < 4; ++t2) {
        const unsigned short* vp = sV + (t2 * 16 + l16) * 64;
        Frag vf0, vf1;
        vf0.u[0] = *(const uint4*)(vp + half * 8);
        vf0.u[1] = *(const uint4*)(vp + 16 + half * 8);
        vf1.u[0] = *(const uint4*)(vp + 32 + half * 8);
        vf1.u[1] = *(const uint4*)(vp + 48 + half * 8);
        oacc[t2] = WMMA_BF16(pf0.v, vf0.v, oacc[t2]);
        oacc[t2] = WMMA_BF16(pf1.v, vf1.v, oacc[t2]);
      }
    }
    __syncthreads();
  }

  // normalize + merge heads: O[b*2048+q][h*64 + d]  (bf16)
#pragma unroll
  for (int t = 0; t < 4; ++t) {
#pragma unroll
    for (int i = 0; i < 8; ++i) {
      const int qr = qbase + half * 8 + i;
      const float o = oacc[t][i] / lrow[i];
      Ob[(size_t)(b * 2048 + qr) * 1024 + h * 64 + t * 16 + l16] = f32_to_bf16(o);
    }
  }
}

// ---------------------------------------------------------------------------
// Host launcher
// ---------------------------------------------------------------------------
extern "C" void kernel_launch(void* const* d_in, const int* in_sizes, int n_in,
                              void* d_out, int out_size, void* d_ws, size_t ws_size,
                              hipStream_t stream) {
  (void)in_sizes; (void)n_in; (void)out_size; (void)ws_size;
  const float* x  = (const float*)d_in[0];
  const float* Wa = (const float*)d_in[1];
  const float* ba = (const float*)d_in[2];
  const float* Wp = (const float*)d_in[3];
  const float* bp = (const float*)d_in[4];
  float* out = (float*)d_out;

  const size_t M = 8192, D = 1024, N3 = 3072;

  // workspace carve-up (bf16 elements): 88 MiB total
  unsigned short* ws  = (unsigned short*)d_ws;
  unsigned short* xb  = ws;              // [M][D]        16 MiB
  unsigned short* WtA = xb  + M * D;     // [3D][D]        6 MiB
  unsigned short* WtP = WtA + N3 * D;    // [D][D]         2 MiB
  unsigned short* Qb  = WtP + D * D;     // [B*H][S][64]  16 MiB (pre-scaled)
  unsigned short* Kb  = Qb  + M * D;     // [B*H][S][64]  16 MiB
  unsigned short* Vtb = Kb  + M * D;     // [B*H][64][S]  16 MiB
  unsigned short* Ob  = Vtb + M * D;     // [M][D]        16 MiB

  cvt_f32_bf16_kernel<<<(int)((M * D) / 1024), 256, 0, stream>>>(x, xb, (int)(M * D));
  transpose_cvt_kernel<<<dim3((unsigned)(N3 / 32), (unsigned)(D / 32)), dim3(32, 8), 0, stream>>>(
      Wa, WtA, (int)D, (int)N3);
  transpose_cvt_kernel<<<dim3((unsigned)(D / 32), (unsigned)(D / 32)), dim3(32, 8), 0, stream>>>(
      Wp, WtP, (int)D, (int)D);

  gemm_bf16_kernel<<<dim3((unsigned)(M / 128), (unsigned)(N3 / 128)), 256, 0, stream>>>(
      xb, WtA, ba, (int)D, 0, Qb, Kb, Vtb, nullptr, (int)N3);

  flash_attn_kernel<<<dim3(64, 16), 256, 0, stream>>>(Qb, Kb, Vtb, Ob);

  gemm_bf16_kernel<<<dim3((unsigned)(M / 128), (unsigned)(D / 128)), 256, 0, stream>>>(
      Ob, WtP, bp, (int)D, 1, nullptr, nullptr, nullptr, out, (int)D);
}
